// AutoCorrelation_20633022890489
// MI455X (gfx1250) — matrix operand order repaired and checked by reading, not compiled, not run on hardware
//
#include <hip/hip_runtime.h>
#include <hip/hip_bf16.h>
#include <math.h>

#define B_   16
#define L_   4096
#define D_   512
#define TOPK 5

typedef __attribute__((ext_vector_type(16))) __bf16 v16bf;
typedef __attribute__((ext_vector_type(8)))  __bf16 v8bf;
typedef __attribute__((ext_vector_type(2)))  __bf16 v2bf;
typedef __attribute__((ext_vector_type(8)))  float  v8f;

union Frag16 {
  v16bf v;
  v8bf  h2[2];
  v2bf  h8[8];
};

// Packed f32 -> bf16 (RNE). Prefer the hardware packed cvt when the builtin
// exists; otherwise let clang lower the scalar fptrunc (also RNE).
__device__ __forceinline__ v2bf pk2(float a, float b) {
#if __has_builtin(__builtin_amdgcn_cvt_pk_bf16_f32)
  return __builtin_amdgcn_cvt_pk_bf16_f32(a, b);
#else
  v2bf r;
  r[0] = (__bf16)a;
  r[1] = (__bf16)b;
  return r;
#endif
}

// ---------------------------------------------------------------------------
// Phase 0a: zero the per-batch score accumulators in workspace
// ---------------------------------------------------------------------------
__global__ void ac_zero_kernel(float* __restrict__ p, int n) {
  int i = blockIdx.x * 256 + threadIdx.x;
  if (i < n) p[i] = 0.0f;
}

// ---------------------------------------------------------------------------
// Phase 0b: bulk f32 -> bf16 conversion (8 elements / thread, b128 in, b128 out)
// ---------------------------------------------------------------------------
__global__ __launch_bounds__(256) void ac_cvt_kernel(
    const float* __restrict__ src, __bf16* __restrict__ dst, int n8) {
  int i = blockIdx.x * 256 + threadIdx.x;
  if (i >= n8) return;
  const float4* s = (const float4*)src;
  float4 a = s[2 * i];
  float4 b = s[2 * i + 1];
  Frag16 u;
  u.h8[0] = pk2(a.x, a.y);
  u.h8[1] = pk2(a.z, a.w);
  u.h8[2] = pk2(b.x, b.y);
  u.h8[3] = pk2(b.z, b.w);
  ((v8bf*)dst)[i] = u.h2[0];
}

// ---------------------------------------------------------------------------
// Shared diagonal fold: C tile (i0..i0+15) x (j0..j0+15) into scoreAcc[tau],
// tau = (i - j) mod L.  C layout: VGPR v, lanes<16: M=v,N=lane; lanes>=16:
// M=v+8, N=lane-16  =>  N == r, M = v + 8*hh.
// ---------------------------------------------------------------------------
__device__ __forceinline__ void fold_tile(float* scoreAcc, v8f acc,
                                          int i0, int j0, int r, int hh) {
  const int base = i0 - j0 - r + 8 * hh;
#pragma unroll
  for (int v = 0; v < 8; ++v) {
    int tau = (base + v) & (L_ - 1);
    atomicAdd(&scoreAcc[tau], acc[v]);
  }
}

// ---------------------------------------------------------------------------
// Phase 1 (fast path): Gram + diagonal fold from pre-converted bf16 Q/K.
// Grid: (L/64, B). Block: 256 = 8 waves. Wave w owns i-tile (w%4), j-tiles
// jt = w/4 + 2t. Inner loop: 2x global_load_b128 + 1 WMMA per 32-wide k-chunk.
// ---------------------------------------------------------------------------
__global__ __launch_bounds__(256) void ac_gram_bf16_kernel(
    const __bf16* __restrict__ Qb, const __bf16* __restrict__ Kb,
    float* __restrict__ scores) {
  __shared__ float scoreAcc[L_];
  const int tid = threadIdx.x;
  for (int i = tid; i < L_; i += 256) scoreAcc[i] = 0.0f;
  __syncthreads();

  const int b    = blockIdx.y;
  const int wave = tid >> 5;
  const int lane = tid & 31;
  const int hh   = lane >> 4;     // half-wave select (ISA 16-bit A/B layout)
  const int r    = lane & 15;
  const int i0   = blockIdx.x * 64 + (wave & 3) * 16;
  const size_t bbase = (size_t)b * L_ * D_;

  // A 16x32 bf16 layout: lane<16 row r holds K={c..c+7, c+16..c+23};
  //                      lane>=16 row r holds K={c+8..c+15, c+24..c+31}.
  const __bf16* qrow = Qb + bbase + (size_t)(i0 + r) * D_;
  v16bf afrag[16];
#pragma unroll
  for (int c = 0; c < 16; ++c) {
    Frag16 f;
    f.h2[0] = *(const v8bf*)(qrow + c * 32 + 8 * hh);
    f.h2[1] = *(const v8bf*)(qrow + c * 32 + 8 * hh + 16);
    afrag[c] = f.v;
  }

  for (int jt = (wave >> 2); jt < (L_ / 16); jt += 2) {
    const int j0 = jt * 16;
    // B 32x16 bf16 layout: lane holds column r, K = {8*hh..8*hh+7, 8*hh+16..8*hh+23}.
    const __bf16* krow = Kb + bbase + (size_t)(j0 + r) * D_ + 8 * hh;
    v8f acc = {};
#pragma unroll
    for (int c = 0; c < 16; ++c) {
      Frag16 f;
      f.h2[0] = *(const v8bf*)(krow + c * 32);
      f.h2[1] = *(const v8bf*)(krow + c * 32 + 16);
      acc = __builtin_amdgcn_wmma_f32_16x16x32_bf16(
          false, afrag[c], false, f.v, (short)0, acc, false, false);
    }
    fold_tile(scoreAcc, acc, i0, j0, r, hh);
  }

  __syncthreads();
  float* gs = scores + (size_t)b * L_;
  for (int i = tid; i < L_; i += 256) atomicAdd(&gs[i], scoreAcc[i]);
}

// ---------------------------------------------------------------------------
// Phase 1 (fallback path, small ws): convert f32 -> bf16 on the fly with the
// packed cvt; same structure otherwise.
// ---------------------------------------------------------------------------
__global__ __launch_bounds__(256) void ac_gram_f32_kernel(
    const float* __restrict__ Q, const float* __restrict__ K,
    float* __restrict__ scores) {
  __shared__ float scoreAcc[L_];
  const int tid = threadIdx.x;
  for (int i = tid; i < L_; i += 256) scoreAcc[i] = 0.0f;
  __syncthreads();

  const int b    = blockIdx.y;
  const int wave = tid >> 5;
  const int lane = tid & 31;
  const int hh   = lane >> 4;
  const int r    = lane & 15;
  const int i0   = blockIdx.x * 64 + (wave & 3) * 16;
  const size_t bbase = (size_t)b * L_ * D_;

  const float* qrow = Q + bbase + (size_t)(i0 + r) * D_;
  v16bf afrag[16];
#pragma unroll
  for (int c = 0; c < 16; ++c) {
    const float* p0 = qrow + c * 32 + 8 * hh;
    const float* p1 = p0 + 16;
    float4 x0 = *(const float4*)(p0);
    float4 x1 = *(const float4*)(p0 + 4);
    float4 x2 = *(const float4*)(p1);
    float4 x3 = *(const float4*)(p1 + 4);
    Frag16 f;
    f.h8[0] = pk2(x0.x, x0.y); f.h8[1] = pk2(x0.z, x0.w);
    f.h8[2] = pk2(x1.x, x1.y); f.h8[3] = pk2(x1.z, x1.w);
    f.h8[4] = pk2(x2.x, x2.y); f.h8[5] = pk2(x2.z, x2.w);
    f.h8[6] = pk2(x3.x, x3.y); f.h8[7] = pk2(x3.z, x3.w);
    afrag[c] = f.v;
  }

  for (int jt = (wave >> 2); jt < (L_ / 16); jt += 2) {
    const int j0 = jt * 16;
    const float* krow = K + bbase + (size_t)(j0 + r) * D_ + 8 * hh;
    v8f acc = {};
#pragma unroll
    for (int c = 0; c < 16; ++c) {
      const float* p = krow + c * 32;
      float4 y0 = *(const float4*)(p);
      float4 y1 = *(const float4*)(p + 4);
      float4 y2 = *(const float4*)(p + 16);
      float4 y3 = *(const float4*)(p + 20);
      Frag16 f;
      f.h8[0] = pk2(y0.x, y0.y); f.h8[1] = pk2(y0.z, y0.w);
      f.h8[2] = pk2(y1.x, y1.y); f.h8[3] = pk2(y1.z, y1.w);
      f.h8[4] = pk2(y2.x, y2.y); f.h8[5] = pk2(y2.z, y2.w);
      f.h8[6] = pk2(y3.x, y3.y); f.h8[7] = pk2(y3.z, y3.w);
      acc = __builtin_amdgcn_wmma_f32_16x16x32_bf16(
          false, afrag[c], false, f.v, (short)0, acc, false, false);
    }
    fold_tile(scoreAcc, acc, i0, j0, r, hh);
  }

  __syncthreads();
  float* gs = scores + (size_t)b * L_;
  for (int i = tid; i < L_; i += 256) atomicAdd(&gs[i], scoreAcc[i]);
}

// ---------------------------------------------------------------------------
// Phase 2: per-batch top-5 + softmax (with the reference's 1/D mean scale)
// ---------------------------------------------------------------------------
__global__ __launch_bounds__(256) void ac_topk_softmax_kernel(
    const float* __restrict__ scores, int* __restrict__ shifts,
    float* __restrict__ weights) {
  __shared__ float s[L_];
  __shared__ float rv[256];
  __shared__ int   ri[256];
  __shared__ float topv[TOPK];
  __shared__ int   topi[TOPK];

  const int b = blockIdx.x;
  const int tid = threadIdx.x;
  const float inv_d = 1.0f / (float)D_;

  for (int i = tid; i < L_; i += 256) s[i] = scores[(size_t)b * L_ + i] * inv_d;
  __syncthreads();

  for (int k = 0; k < TOPK; ++k) {
    float best = -INFINITY; int bi = 0;
    for (int i = tid; i < L_; i += 256) {
      float v = s[i];
      if (v > best) { best = v; bi = i; }
    }
    rv[tid] = best; ri[tid] = bi;
    __syncthreads();
    for (int off = 128; off > 0; off >>= 1) {
      if (tid < off && rv[tid + off] > rv[tid]) {
        rv[tid] = rv[tid + off]; ri[tid] = ri[tid + off];
      }
      __syncthreads();
    }
    if (tid == 0) {
      topv[k] = rv[0]; topi[k] = ri[0];
      s[ri[0]] = -INFINITY;
    }
    __syncthreads();
  }

  if (tid == 0) {
    float m = topv[0];
    float e[TOPK], sum = 0.0f;
    for (int k = 0; k < TOPK; ++k) { e[k] = expf(topv[k] - m); sum += e[k]; }
    for (int k = 0; k < TOPK; ++k) {
      weights[b * TOPK + k] = e[k] / sum;
      shifts[b * TOPK + k]  = topi[k];
    }
  }
}

// ---------------------------------------------------------------------------
// Phase 3: out[b,l,:] = sum_k w_k * V[b, (l - shift_k) % L, :]  (float4 vec)
// ---------------------------------------------------------------------------
__global__ __launch_bounds__(256) void ac_gather_kernel(
    const float* __restrict__ V, const int* __restrict__ shifts,
    const float* __restrict__ weights, float* __restrict__ out) {
  const int b  = blockIdx.y;
  const int t  = blockIdx.x * 256 + threadIdx.x;  // over L * (D/4)
  const int l  = t >> 7;                          // D/4 == 128
  const int d4 = t & 127;

  int   sh[TOPK];
  float w[TOPK];
#pragma unroll
  for (int k = 0; k < TOPK; ++k) {
    sh[k] = shifts[b * TOPK + k];
    w[k]  = weights[b * TOPK + k];
  }

  float4 acc = make_float4(0.f, 0.f, 0.f, 0.f);
#pragma unroll
  for (int k = 0; k < TOPK; ++k) {
    int src = (l - sh[k]) & (L_ - 1);
    const float4* vp = (const float4*)(V + ((size_t)b * L_ + src) * D_);
    float4 x = vp[d4];
    acc.x += w[k] * x.x; acc.y += w[k] * x.y;
    acc.z += w[k] * x.z; acc.w += w[k] * x.w;
  }
  float4* op = (float4*)(out + ((size_t)b * L_ + l) * D_);
  op[d4] = acc;
}

// ---------------------------------------------------------------------------
extern "C" void kernel_launch(void* const* d_in, const int* in_sizes, int n_in,
                              void* d_out, int out_size, void* d_ws, size_t ws_size,
                              hipStream_t stream) {
  (void)in_sizes; (void)n_in; (void)out_size;
  const float* Q = (const float*)d_in[0];
  const float* K = (const float*)d_in[1];
  const float* V = (const float*)d_in[2];
  float* out = (float*)d_out;

  // ws layout: [scores B*L f32][shifts B*K i32 | weights B*K f32][Qbf][Kbf]
  const size_t score_bytes = (size_t)B_ * L_ * sizeof(float);
  float* scores  = (float*)d_ws;
  char*  meta    = (char*)d_ws + score_bytes;
  int*   shifts  = (int*)meta;
  float* weights = (float*)(meta + 256);
  const size_t bf_off   = score_bytes + 512;
  const size_t bf_bytes = (size_t)B_ * L_ * D_ * 2;

  ac_zero_kernel<<<(B_ * L_ + 255) / 256, 256, 0, stream>>>(scores, B_ * L_);

  if (ws_size >= bf_off + 2 * bf_bytes) {
    // Fast path: one-shot bf16 conversion, then pure-WMMA Gram sweep.
    __bf16* Qb = (__bf16*)((char*)d_ws + bf_off);
    __bf16* Kb = (__bf16*)((char*)d_ws + bf_off + bf_bytes);
    const int n8 = B_ * L_ * D_ / 8;
    ac_cvt_kernel<<<(n8 + 255) / 256, 256, 0, stream>>>(Q, Qb, n8);
    ac_cvt_kernel<<<(n8 + 255) / 256, 256, 0, stream>>>(K, Kb, n8);
    ac_gram_bf16_kernel<<<dim3(L_ / 64, B_), 256, 0, stream>>>(Qb, Kb, scores);
  } else {
    // Fallback: convert on the fly inside the Gram kernel.
    ac_gram_f32_kernel<<<dim3(L_ / 64, B_), 256, 0, stream>>>(Q, K, scores);
  }

  ac_topk_softmax_kernel<<<B_, 256, 0, stream>>>(scores, shifts, weights);
  ac_gather_kernel<<<dim3((L_ * (D_ / 4)) / 256, B_), 256, 0, stream>>>(
      V, shifts, weights, out);
}
